// EncoderProcesserDecoder_46231027974469
// MI455X (gfx1250) — matrix operand order repaired and checked by reading, not compiled, not run on hardware
//
#include <hip/hip_runtime.h>
#include <hip/hip_bf16.h>

// ---------------------------------------------------------------------------
// MeshGraphNet encode-process-decode for MI455X (gfx1250, wave32, WMMA+TDM).
//
// All per-row MLPs are tall-skinny GEMMs done with v_wmma_f32_16x16x32_bf16
// (f32 accumulate). Activations stored bf16 (halves HBM traffic; 23.3 TB/s).
// Each MLP (3 layers + LayerNorm + residual + gather/scatter) is fused into
// one kernel; intermediates live in LDS (~145KB of the 320KB WGP budget).
// Weight chunks are staged LDS-side by the Tensor Data Mover
// (tensor_load_to_lds, double-buffered, s_wait_tensorcnt) so the DMA overlaps
// the WMMA math. segment_sum uses global f32 atomics into an f32 workspace.
//
// Input flattening assumption (dict-insertion order, 124 leaves):
//   0 node_attr  1 edge_attr  2 y  3 out_mask
//   4..11   enc_node  {W0(11x128) b0 W1 b1 W2 b2 ln_g ln_b}
//   12..19  enc_edge  {W0(3x128)  b0 W1 b1 W2 b2 ln_g ln_b}
//   20+16b..: block b: eb {W0(384x128) b0 W1 b1 W2 b2 g be}
//                      nb {W0(192x128) b0 W1 b1 W2 b2 g be}
//   116..121 dec {W0(256x128) b0 W1 b1 W2(128x3) b2}
//   122 edge_index (int32, [2][300000])   123 node_type (int32)
// ---------------------------------------------------------------------------

typedef unsigned int uint32;
typedef unsigned long long uint64;
typedef __bf16 bf16;
typedef __attribute__((ext_vector_type(16))) __bf16 v16bf;
typedef __attribute__((ext_vector_type(8)))  float  v8f;
typedef __attribute__((ext_vector_type(4)))  unsigned int u32x4;
typedef __attribute__((ext_vector_type(8)))  int          i32x8;
typedef __attribute__((ext_vector_type(4)))  int          i32x4;

constexpr int HIDDEN  = 128;
constexpr int NNODES  = 100000;
constexpr int NEDGES  = 300000;
constexpr int ROWS    = 64;    // rows per workgroup tile
constexpr int THREADS = 256;   // 8 waves of 32
constexpr int WBUF    = 128 * 32;  // dwords per weight staging buffer (16KB)

#if defined(__has_builtin)
#if __has_builtin(__builtin_amdgcn_tensor_load_to_lds) && __has_builtin(__builtin_amdgcn_s_wait_tensorcnt)
#define HAVE_TDM 1
#endif
#endif

struct MlpArgs {
  const uint32* w0; const uint32* w1; const uint32* w2;   // packed bf16 weights [128][K/2] dwords
  const float* b0; const float* b1; const float* b2;
  const float* ln_g; const float* ln_b;
  const float* attr;           // encoder raw input (f32)
  const int*   eidx;           // edge_index (mode 2)
  bf16* xBuf; bf16* eBuf; bf16* embBuf;
  float* agg;
  const float* decW2; const float* decB2;                  // decoder last layer (128x3, f32)
  const float* y; const float* outMask;
  const int* ntype;
  float* out;
  int nRowsTotal;
};

__device__ __forceinline__ float silu_fast(float x) {
  // x * sigmoid(x) with v_exp_f32 + v_rcp_f32 (avoid IEEE divide expansion)
  return x * __builtin_amdgcn_rcpf(1.0f + __expf(-x));
}

#ifdef HAVE_TDM
// Issue one TDM 2D-tile load: 128 rows x pcDwords dwords, global row stride
// strideDwords, compacted into LDS at sWbuf. Per-wave op, EXEC ignored.
// 6-arg builtin form: (g0 u32x4, g1 i32x8, g2 i32x4, g3 i32x4, i32x8, cpol).
__device__ __forceinline__ void tdm_load_chunk(uint32* sWbuf, const uint32* gsrc,
                                               uint32 strideDwords, uint32 pcDwords)
{
  uint64 ga  = (uint64)(uintptr_t)gsrc;                // global byte address
  uint32 lds = (uint32)(uintptr_t)sWbuf;               // low 32 bits = LDS offset
  u32x4 g0;
  g0.x = 1u;                                           // count=1 (valid user D#)
  g0.y = lds;                                          // lds_addr
  g0.z = (uint32)ga;                                   // global_addr[31:0]
  g0.w = (uint32)(ga >> 32) | (2u << 30);              // global_addr[56:32] | type=2
  i32x8 g1;
  g1[0] = (int)(2u << 16);                             // wg_mask=0 | data_size=4B
  g1[1] = (int)((strideDwords & 0xFFFFu) << 16);       // tensor_dim0 lo16
  g1[2] = (int)((strideDwords >> 16) | (128u << 16));  // tensor_dim0 hi | tensor_dim1 lo (=128)
  g1[3] = (int)(pcDwords << 16);                       // tensor_dim1 hi (=0) | tile_dim0
  g1[4] = (int)128u;                                   // tile_dim1=128 | tile_dim2=0
  g1[5] = (int)strideDwords;                           // tensor_dim0_stride lo32
  g1[6] = 0;                                           // stride hi | dim1_stride lo
  g1[7] = 0;
  i32x4 z4 = {0, 0, 0, 0};                             // groups 2/3 unused (2D)
  i32x8 z8 = {0, 0, 0, 0, 0, 0, 0, 0};
  __builtin_amdgcn_tensor_load_to_lds(g0, g1, z4, z4, z8, 0);
}
#endif

// ---------------------------------------------------------------------------
// One GEMM layer: out[64][128] = act(sIn[64][KINL] @ W + b)
// A frag (16x32 bf16): lane m = row, lane-half h holds k 8h..8h+7 and 16+8h..16+8h+7 -> two 16B LDS runs.
// B frag (32x16 bf16): weights pre-packed {k_even,k_odd} dwords, [n][k/2]
// layout -> two contiguous 16B runs per lane. Weight chunks double-buffered
// in LDS via TDM so DMA overlaps WMMA.
// ---------------------------------------------------------------------------
template<int KINL, bool ACT, bool OUTF32>
__device__ __forceinline__ void mlp_layer(const bf16* __restrict__ sIn,
                                          const uint32* __restrict__ gW,
                                          const float* __restrict__ bias,
                                          uint32* __restrict__ sW,   // 2*WBUF dwords
                                          bf16* __restrict__ outB,
                                          float* __restrict__ outF)
{
  constexpr int CH  = (KINL >= 64) ? 64 : KINL;   // K rows per staged chunk
  constexpr int PC  = CH / 2;                     // dword pairs per chunk row
  constexpr int NCH = KINL / CH;

  const int tid     = threadIdx.x;
  const int lane    = tid & 31;
  const int wave    = tid >> 5;
  const int rowBase = (wave & 3) * 16;            // 4 row bands
  const int colBase = (wave >> 2) * 64;           // 2 column halves
  const int m       = lane & 15;
  const int lh      = lane >> 4;

  v8f acc[4] = {};

#ifdef HAVE_TDM
  if (wave == 0) tdm_load_chunk(sW, gW, KINL / 2, PC);
#endif

  for (int c = 0; c < NCH; ++c) {
    uint32* cur = sW + (c & 1) * WBUF;
#ifdef HAVE_TDM
    if (wave == 0) {
      if (c + 1 < NCH) {
        tdm_load_chunk(sW + ((c + 1) & 1) * WBUF, gW + (size_t)(c + 1) * PC,
                       KINL / 2, PC);
        __builtin_amdgcn_s_wait_tensorcnt(1);   // chunk c landed, c+1 in flight
      } else {
        __builtin_amdgcn_s_wait_tensorcnt(0);
      }
    }
    __syncthreads();
#else
    __syncthreads();
    for (int q = tid; q < 128 * PC / 4; q += THREADS) {
      int n  = q / (PC / 4);
      int po = (q % (PC / 4)) * 4;
      *reinterpret_cast<uint4*>(&cur[n * PC + po]) =
        *reinterpret_cast<const uint4*>(&gW[(size_t)n * (KINL / 2) + (size_t)c * PC + po]);
    }
    __syncthreads();
#endif
    #pragma unroll
    for (int ks = 0; ks < CH / 32; ++ks) {
      const bf16* pa = sIn + (rowBase + m) * KINL + c * CH + ks * 32 + lh * 8;
      v16bf a;
      reinterpret_cast<uint4*>(&a)[0] = *reinterpret_cast<const uint4*>(pa);
      reinterpret_cast<uint4*>(&a)[1] = *reinterpret_cast<const uint4*>(pa + 16);
      #pragma unroll
      for (int j = 0; j < 4; ++j) {
        const uint32* pb = cur + (colBase + j * 16 + m) * PC + ks * 16 + lh * 4;
        v16bf b;
        reinterpret_cast<uint4*>(&b)[0] = *reinterpret_cast<const uint4*>(pb);
        reinterpret_cast<uint4*>(&b)[1] = *reinterpret_cast<const uint4*>(pb + 8);
        acc[j] = __builtin_amdgcn_wmma_f32_16x16x32_bf16(
                   false, a, false, b, (short)0, acc[j], false, false);
      }
    }
    __syncthreads();   // all waves done with `cur` before it is refilled
  }

  // epilogue (branch-free): bias + SiLU, C/D layout row = v + 8*lh, col = m
  #pragma unroll
  for (int j = 0; j < 4; ++j) {
    const int col  = colBase + j * 16 + m;
    const float bv = bias[col];
    #pragma unroll
    for (int v = 0; v < 8; ++v) {
      float x = acc[j][v] + bv;
      if constexpr (ACT) x = silu_fast(x);
      const int row = rowBase + lh * 8 + v;
      if constexpr (OUTF32) outF[row * 128 + col] = x;
      else                  outB[row * 128 + col] = (bf16)x;
    }
  }
}

// MODE: 0=encode nodes, 1=encode edges, 2=edge block (gather+LN+scatter+res),
//       3=node block (LN+res), 4=decoder (+postprocess)
template<int KIN, int MODE>
__global__ void __launch_bounds__(THREADS)
mlp_kernel(MlpArgs a)
{
  __shared__ __align__(16) bf16   sX [ROWS * KIN];
  __shared__ __align__(16) uint32 sW [2 * WBUF];       // double-buffered weights
  __shared__ __align__(16) bf16   sH1[ROWS * 128];
  __shared__ __align__(16) bf16   sH2[ROWS * 128];
  __shared__ __align__(16) float  sF [ROWS * 128];
  __shared__ int   sS[ROWS];
  __shared__ int   sR[ROWS];
  __shared__ float sMu[ROWS];
  __shared__ float sRstd[ROWS];

  const int tid      = threadIdx.x;
  const int tileBase = blockIdx.x * ROWS;

  // ---------------- gather input tile into sX (bf16) ----------------
  if constexpr (MODE == 0 || MODE == 1) {
    constexpr int AC = (MODE == 0) ? 11 : 3;   // raw attr cols, zero-pad to 32
    for (int idx = tid; idx < ROWS * KIN; idx += THREADS) {
      int row = idx / KIN, col = idx % KIN;
      int g = tileBase + row;
      float v = 0.0f;
      if (g < a.nRowsTotal && col < AC) v = a.attr[(size_t)g * AC + col];
      sX[idx] = (bf16)v;
    }
  }
  if constexpr (MODE == 2) {   // concat([e, x[s], x[r]])
    if (tid < ROWS) {
      int g = tileBase + tid;
      int s = 0, r = 0;
      if (g < a.nRowsTotal) { s = a.eidx[g]; r = a.eidx[NEDGES + g]; }
      sS[tid] = s; sR[tid] = r;
    }
    __syncthreads();
    for (int seg = tid; seg < ROWS * 3; seg += THREADS) {
      int row = seg / 3, part = seg % 3;
      int g = tileBase + row;
      uint4* dst = reinterpret_cast<uint4*>(sX + row * KIN + part * 128);
      if (g < a.nRowsTotal) {
        const bf16* srcp = (part == 0) ? (a.eBuf + (size_t)g * 128)
                         : (part == 1) ? (a.xBuf + (size_t)sS[row] * 128)
                                       : (a.xBuf + (size_t)sR[row] * 128);
        const uint4* src = reinterpret_cast<const uint4*>(srcp);
        #pragma unroll
        for (int i = 0; i < 16; ++i) dst[i] = src[i];
      } else {
        uint4 z = {0, 0, 0, 0};
        #pragma unroll
        for (int i = 0; i < 16; ++i) dst[i] = z;
      }
    }
  }
  if constexpr (MODE == 3) {   // concat([x, agg])
    for (int q = tid; q < ROWS * 16; q += THREADS) {
      int row = q >> 4, i = q & 15;
      int g = tileBase + row;
      uint4* dst = reinterpret_cast<uint4*>(sX + row * KIN) + i;
      if (g < a.nRowsTotal)
        *dst = reinterpret_cast<const uint4*>(a.xBuf + (size_t)g * 128)[i];
      else { uint4 z = {0, 0, 0, 0}; *dst = z; }
    }
    for (int q = tid; q < ROWS * 64; q += THREADS) {
      int row = q >> 6, c = q & 63;
      int g = tileBase + row;
      float v = (g < a.nRowsTotal) ? a.agg[(size_t)g * 64 + c] : 0.0f;
      sX[row * KIN + 128 + c] = (bf16)v;
    }
  }
  if constexpr (MODE == 4) {   // concat([x, node_embedding])
    for (int seg = tid; seg < ROWS * 2; seg += THREADS) {
      int row = seg >> 1, part = seg & 1;
      int g = tileBase + row;
      uint4* dst = reinterpret_cast<uint4*>(sX + row * KIN + part * 128);
      if (g < a.nRowsTotal) {
        const bf16* srcp = (part == 0) ? (a.xBuf + (size_t)g * 128)
                                       : (a.embBuf + (size_t)g * 128);
        const uint4* src = reinterpret_cast<const uint4*>(srcp);
        #pragma unroll
        for (int i = 0; i < 16; ++i) dst[i] = src[i];
      } else {
        uint4 z = {0, 0, 0, 0};
        #pragma unroll
        for (int i = 0; i < 16; ++i) dst[i] = z;
      }
    }
  }

  // ---------------- fused 3-layer MLP via WMMA ----------------
  __syncthreads();     // sX visible before first TDM-overlapped compute
  mlp_layer<KIN, true, false>(sX,  a.w0, a.b0, sW, sH1, nullptr);
  mlp_layer<128, true, false>(sH1, a.w1, a.b1, sW, sH2, nullptr);

  if constexpr (MODE != 4) {
    mlp_layer<128, false, true>(sH2, a.w2, a.b2, sW, nullptr, sF);
    __syncthreads();
    // LayerNorm stats, one row per thread
    if (tid < ROWS) {
      float s = 0.0f, ss = 0.0f;
      #pragma unroll 4
      for (int c = 0; c < 128; ++c) { float v = sF[tid * 128 + c]; s += v; ss += v * v; }
      float mu  = s * (1.0f / 128.0f);
      float var = ss * (1.0f / 128.0f) - mu * mu;
      sMu[tid]   = mu;
      sRstd[tid] = rsqrtf(var + 1e-5f);
    }
    __syncthreads();
    // LN apply + residual + scatter
    for (int idx = tid; idx < ROWS * 128; idx += THREADS) {
      int row = idx >> 7, col = idx & 127;
      int g = tileBase + row;
      if (g >= a.nRowsTotal) continue;
      float v = (sF[idx] - sMu[row]) * sRstd[row] * a.ln_g[col] + a.ln_b[col];
      if constexpr (MODE == 0) {
        bf16 o = (bf16)v;
        a.xBuf  [(size_t)g * 128 + col] = o;
        a.embBuf[(size_t)g * 128 + col] = o;
      } else if constexpr (MODE == 1) {
        a.eBuf[(size_t)g * 128 + col] = (bf16)v;
      } else if constexpr (MODE == 2) {
        float eo = (float)sX[row * KIN + col];          // old e (residual)
        a.eBuf[(size_t)g * 128 + col] = (bf16)(eo + v); // e += e_new
        if (col < 64) atomicAdd(&a.agg[(size_t)sR[row] * 64 + col], v);
        else          atomicAdd(&a.agg[(size_t)sS[row] * 64 + (col - 64)], v);
      } else { // MODE == 3
        float xo = (float)sX[row * KIN + col];          // old x (residual)
        a.xBuf[(size_t)g * 128 + col] = (bf16)(xo + v); // x += x_new
      }
    }
  } else {
    // decoder tail: 128 -> 3 (VALU), then postprocess
    __syncthreads();
    if (tid < ROWS * 3) {
      int row = tid / 3, o = tid % 3;
      int g = tileBase + row;
      if (g < a.nRowsTotal) {
        float acc = a.decB2[o];
        for (int c2 = 0; c2 < 128; ++c2)
          acc += (float)sH2[row * 128 + c2] * a.decW2[c2 * 3 + o];
        sF[row * 3 + o] = acc;
      }
    }
    __syncthreads();
    if (tid < ROWS) {
      int g = tileBase + tid;
      if (g < a.nRowsTotal) {
        float d0 = sF[tid * 3 + 0], d1 = sF[tid * 3 + 1], d2 = sF[tid * 3 + 2];
        float uv0 = 10.0f * tanhf(d0 * 0.1f);
        float uv1 = 10.0f * tanhf(d1 * 0.1f);
        float p   = 10.0f * tanhf(d2 * 0.1f);
        int nt = a.ntype[g];
        if (nt == 4 || nt == 6 || nt == 7) { uv0 = a.y[2 * g]; uv1 = a.y[2 * g + 1]; }
        if (nt == 5 || nt == 7) p = 0.0f;
        a.out[3 * g + 0] = uv0 * a.outMask[3 * g + 0];
        a.out[3 * g + 1] = uv1 * a.outMask[3 * g + 1];
        a.out[3 * g + 2] = p   * a.outMask[3 * g + 2];
      }
    }
  }
}

// Pack f32 weights [K][128] -> bf16 pair dwords, transposed: out[n][Kpad/2],
// dword = {bf16 W[2p][n] (lo), bf16 W[2p+1][n] (hi)}; zero-pad K -> Kpad.
__global__ void pack_weight_kernel(const float* __restrict__ W,
                                   uint32* __restrict__ out, int K, int Kpad)
{
  int pp  = Kpad >> 1;
  int idx = blockIdx.x * blockDim.x + threadIdx.x;
  if (idx >= 128 * pp) return;
  int n = idx / pp;
  int p = idx - n * pp;
  int k0 = 2 * p, k1 = k0 + 1;
  float w0 = (k0 < K) ? W[(size_t)k0 * 128 + n] : 0.0f;
  float w1 = (k1 < K) ? W[(size_t)k1 * 128 + n] : 0.0f;
  bf16 h0 = (bf16)w0, h1 = (bf16)w1;
  unsigned short u0, u1;
  __builtin_memcpy(&u0, &h0, 2);
  __builtin_memcpy(&u1, &h1, 2);
  out[idx] = (uint32)u0 | ((uint32)u1 << 16);
}

__global__ void zero_kernel(float* p, int n)
{
  int i = blockIdx.x * blockDim.x + threadIdx.x;
  if (i < n) p[i] = 0.0f;
}

extern "C" void kernel_launch(void* const* d_in, const int* in_sizes, int n_in,
                              void* d_out, int out_size, void* d_ws, size_t ws_size,
                              hipStream_t stream)
{
  (void)in_sizes; (void)n_in; (void)out_size; (void)ws_size;

  const float* node_attr = (const float*)d_in[0];
  const float* edge_attr = (const float*)d_in[1];
  const float* y         = (const float*)d_in[2];
  const float* out_mask  = (const float*)d_in[3];
  const int*   edge_index= (const int*)d_in[122];   // assumed int32
  const int*   node_type = (const int*)d_in[123];

  // ---------------- workspace layout ----------------
  char* ws = (char*)d_ws;
  size_t off = 0;
  bf16*  xBuf   = (bf16*)(ws + off); off += (size_t)NNODES * 128 * 2;
  bf16*  embBuf = (bf16*)(ws + off); off += (size_t)NNODES * 128 * 2;
  bf16*  eBuf   = (bf16*)(ws + off); off += (size_t)NEDGES * 128 * 2;
  float* agg    = (float*)(ws + off); off += (size_t)NNODES * 64 * 4;
  uint32* wpk   = (uint32*)(ws + off);

  // ---------------- pack all weight matrices ----------------
  struct PW { int idx; int K; int Kpad; };
  PW list[44]; int nl = 0;
  list[nl++] = {4, 11, 32};  list[nl++] = {6, 128, 128};  list[nl++] = {8, 128, 128};   // enc_node
  list[nl++] = {12, 3, 32};  list[nl++] = {14, 128, 128}; list[nl++] = {16, 128, 128};  // enc_edge
  for (int b = 0; b < 6; ++b) {
    int base = 20 + 16 * b;
    list[nl++] = {base + 0, 384, 384}; list[nl++] = {base + 2, 128, 128}; list[nl++] = {base + 4, 128, 128};   // eb
    list[nl++] = {base + 8, 192, 192}; list[nl++] = {base + 10, 128, 128}; list[nl++] = {base + 12, 128, 128}; // nb
  }
  list[nl++] = {116, 256, 256}; list[nl++] = {118, 128, 128};                            // dec W0,W1

  size_t woff[44]; size_t wacc = 0;
  for (int i = 0; i < nl; ++i) { woff[i] = wacc; wacc += (size_t)128 * (list[i].Kpad / 2); }
  for (int i = 0; i < nl; ++i) {
    int total = 128 * (list[i].Kpad / 2);
    pack_weight_kernel<<<(total + 255) / 256, 256, 0, stream>>>(
        (const float*)d_in[list[i].idx], wpk + woff[i], list[i].K, list[i].Kpad);
  }

  const int nodeBlocks = (NNODES + ROWS - 1) / ROWS;
  const int edgeBlocks = (NEDGES + ROWS - 1) / ROWS;

  // ---------------- encoders ----------------
  {
    MlpArgs a = {};
    a.w0 = wpk + woff[0]; a.w1 = wpk + woff[1]; a.w2 = wpk + woff[2];
    a.b0 = (const float*)d_in[5];  a.b1 = (const float*)d_in[7];  a.b2 = (const float*)d_in[9];
    a.ln_g = (const float*)d_in[10]; a.ln_b = (const float*)d_in[11];
    a.attr = node_attr; a.xBuf = xBuf; a.embBuf = embBuf;
    a.nRowsTotal = NNODES;
    mlp_kernel<32, 0><<<nodeBlocks, THREADS, 0, stream>>>(a);
  }
  {
    MlpArgs a = {};
    a.w0 = wpk + woff[3]; a.w1 = wpk + woff[4]; a.w2 = wpk + woff[5];
    a.b0 = (const float*)d_in[13]; a.b1 = (const float*)d_in[15]; a.b2 = (const float*)d_in[17];
    a.ln_g = (const float*)d_in[18]; a.ln_b = (const float*)d_in[19];
    a.attr = edge_attr; a.eBuf = eBuf;
    a.nRowsTotal = NEDGES;
    mlp_kernel<32, 1><<<edgeBlocks, THREADS, 0, stream>>>(a);
  }

  // ---------------- message-passing blocks ----------------
  for (int b = 0; b < 6; ++b) {
    int li   = 6 + 6 * b;
    int base = 20 + 16 * b;
    zero_kernel<<<(NNODES * 64 + 255) / 256, 256, 0, stream>>>(agg, NNODES * 64);
    {
      MlpArgs a = {};
      a.w0 = wpk + woff[li + 0]; a.w1 = wpk + woff[li + 1]; a.w2 = wpk + woff[li + 2];
      a.b0 = (const float*)d_in[base + 1]; a.b1 = (const float*)d_in[base + 3];
      a.b2 = (const float*)d_in[base + 5];
      a.ln_g = (const float*)d_in[base + 6]; a.ln_b = (const float*)d_in[base + 7];
      a.eidx = edge_index; a.xBuf = xBuf; a.eBuf = eBuf; a.agg = agg;
      a.nRowsTotal = NEDGES;
      mlp_kernel<384, 2><<<edgeBlocks, THREADS, 0, stream>>>(a);
    }
    {
      MlpArgs a = {};
      a.w0 = wpk + woff[li + 3]; a.w1 = wpk + woff[li + 4]; a.w2 = wpk + woff[li + 5];
      a.b0 = (const float*)d_in[base + 9]; a.b1 = (const float*)d_in[base + 11];
      a.b2 = (const float*)d_in[base + 13];
      a.ln_g = (const float*)d_in[base + 14]; a.ln_b = (const float*)d_in[base + 15];
      a.xBuf = xBuf; a.agg = agg;
      a.nRowsTotal = NNODES;
      mlp_kernel<192, 3><<<nodeBlocks, THREADS, 0, stream>>>(a);
    }
  }

  // ---------------- decoder + postprocess ----------------
  {
    MlpArgs a = {};
    a.w0 = wpk + woff[42]; a.w1 = wpk + woff[43];
    a.b0 = (const float*)d_in[117]; a.b1 = (const float*)d_in[119];
    a.decW2 = (const float*)d_in[120]; a.decB2 = (const float*)d_in[121];
    a.xBuf = xBuf; a.embBuf = embBuf;
    a.y = y; a.outMask = out_mask; a.ntype = node_type;
    a.out = (float*)d_out;
    a.nRowsTotal = NNODES;
    mlp_kernel<256, 4><<<nodeBlocks, THREADS, 0, stream>>>(a);
  }
}
